// Predictor_23235773071400
// MI455X (gfx1250) — matrix operand order repaired and checked by reading, not compiled, not run on hardware
//
#include <hip/hip_runtime.h>
#include <math.h>

// ---------------------------------------------------------------------------
// Types
// ---------------------------------------------------------------------------
typedef __attribute__((ext_vector_type(16))) __bf16 v16bf;
typedef __attribute__((ext_vector_type(8)))  __bf16 v8bf;
typedef __attribute__((ext_vector_type(8)))  float  v8f;

#define BATCH 4
#define LFULL 19775
#define LQ    16447
#define SFRM  256
#define TTOP  256
#define TBOT  1024
#define TTOK  16384
#define DIMH  512
#define H3    1536

__device__ inline __bf16 f2bf(float f) {
    union { float f; unsigned int u; } v; v.f = f;
    unsigned int r = v.u + 0x7FFFu + ((v.u >> 16) & 1u);   // RNE
    unsigned short h = (unsigned short)(r >> 16);
    __bf16 o; __builtin_memcpy(&o, &h, 2); return o;
}
__device__ inline float bf2f(__bf16 b) {
    unsigned short h; __builtin_memcpy(&h, &b, 2);
    union { unsigned int u; float f; } v; v.u = ((unsigned int)h) << 16; return v.f;
}

// A-matrix fragment (16x32 bf16, MxK): per ISA table, lane half selects K base
// +0 / +8; element chunks live at K+{0..7} and K+{16..23}.
__device__ inline v16bf frag_a(const __bf16* row, int k0, int lane) {
    const __bf16* p = row + k0 + ((lane >> 4) << 3);
    v8bf lo = *(const v8bf*)p;
    v8bf hi = *(const v8bf*)(p + 16);
    return __builtin_shufflevector(lo, hi, 0,1,2,3,4,5,6,7,8,9,10,11,12,13,14,15);
}
// B-matrix fragment (32x16 bf16, KxN): same K order as frag_a -- lane half
// selects K base +0 / +8; element chunks live at K+{0..7} and K+{16..23}.
__device__ inline v16bf frag_b(const __bf16* row, int k0, int lane) {
    const __bf16* p = row + k0 + ((lane >> 4) << 3);
    v8bf lo = *(const v8bf*)p;
    v8bf hi = *(const v8bf*)(p + 16);
    return __builtin_shufflevector(lo, hi, 0,1,2,3,4,5,6,7,8,9,10,11,12,13,14,15);
}
__device__ inline v8f wmma_bf16(v16bf a, v16bf b, v8f c) {
    return __builtin_amdgcn_wmma_f32_16x16x32_bf16(false, a, false, b, (short)0, c,
                                                   false, false);
}

// ---------------------------------------------------------------------------
// Weight conversion
// ---------------------------------------------------------------------------
__global__ __launch_bounds__(256) void k_cvt(const float* __restrict__ s,
                                             __bf16* __restrict__ d, int n) {
    for (int i = blockIdx.x * 256 + threadIdx.x; i < n; i += gridDim.x * 256)
        d[i] = f2bf(s[i]);
}

// Wc (512,128,16) -> WcK (512, 2048) with k = tap*128 + c
__global__ __launch_bounds__(256) void k_cvt_wc(const float* __restrict__ Wc,
                                                __bf16* __restrict__ d) {
    for (int i = blockIdx.x * 256 + threadIdx.x; i < 512 * 2048; i += gridDim.x * 256) {
        int n = i >> 11, k = i & 2047, tap = k >> 7, c = k & 127;
        d[i] = f2bf(Wc[((size_t)n * 128 + c) * 16 + tap]);
    }
}

// ---------------------------------------------------------------------------
// Frame feature extraction: hf (B,256,13)
// ---------------------------------------------------------------------------
__global__ __launch_bounds__(256) void k_features(const int* __restrict__ inp,
                                                  float* __restrict__ hf) {
    int i = blockIdx.x * 256 + threadIdx.x;
    if (i >= BATCH * SFRM * 13) return;
    int j = i % 13, t = (i / 13) % SFRM, b = i / (13 * SFRM);
    const int* row = inp + (size_t)b * LFULL + LQ;
    float v;
    if (j < 10)       v = (float)row[t * 10 + j] * (1.0f / 20000.0f);
    else if (j == 10) v = (float)row[2560 + t] * (1.0f / 650000.0f);
    else if (j == 11) v = (float)row[2816 + t] * (1.0f / 100.0f);
    else              v = (float)row[3072 + t];
    hf[i] = v;
}

// ---------------------------------------------------------------------------
// Top GRU input: x_top (B*256, 512) bf16   (K = 64 samples + 13 feats)
// ---------------------------------------------------------------------------
__global__ __launch_bounds__(256) void k_xtop(const int* __restrict__ inp,
        const float* __restrict__ hf, const float* __restrict__ Wint,
        const float* __restrict__ bint, __bf16* __restrict__ xt) {
    __shared__ float s[64];
    __shared__ float fv[13];
    int bt = blockIdx.x, b = bt >> 8, t = bt & 255, tid = threadIdx.x;
    if (tid < 64) s[tid] = (float)inp[(size_t)b * LFULL + t * 64 + tid] * (1.0f / 64.0f) - 2.0f;
    if (tid >= 64 && tid < 77) fv[tid - 64] = hf[((size_t)b * SFRM + t) * 13 + (tid - 64)];
    __syncthreads();
#pragma unroll
    for (int q = 0; q < 2; q++) {
        int o = tid + q * 256;
        const float* wr = Wint + (size_t)o * 77;
        float a = bint[o];
        for (int i = 0; i < 64; i++) a += s[i] * wr[i];
        for (int j = 0; j < 13; j++) a += fv[j] * wr[64 + j];
        xt[(size_t)bt * DIMH + o] = f2bf(a);
    }
}

// ---------------------------------------------------------------------------
// Bottom GRU input: x_bot = ps@Winb^T + binb + cond_top   (B*1024, 512) bf16
// ---------------------------------------------------------------------------
__global__ __launch_bounds__(256) void k_xbot(const int* __restrict__ inp,
        const float* __restrict__ hf, const float* __restrict__ Winb,
        const float* __restrict__ binb, const float* __restrict__ cond_top,
        __bf16* __restrict__ xb) {
    __shared__ float s[16];
    __shared__ float fv[13];
    int bt = blockIdx.x, b = bt >> 10, t = bt & 1023, tid = threadIdx.x;
    if (tid < 16) s[tid] = (float)inp[(size_t)b * LFULL + 48 + t * 16 + tid] * (1.0f / 64.0f) - 2.0f;
    if (tid >= 16 && tid < 29) fv[tid - 16] = hf[((size_t)b * SFRM + (t >> 2)) * 13 + (tid - 16)];
    __syncthreads();
#pragma unroll
    for (int q = 0; q < 2; q++) {
        int o = tid + q * 256;
        const float* wr = Winb + (size_t)o * 29;
        float a = binb[o];
        for (int i = 0; i < 16; i++) a += s[i] * wr[i];
        for (int j = 0; j < 13; j++) a += fv[j] * wr[16 + j];
        a += cond_top[(size_t)bt * DIMH + o];
        xb[(size_t)bt * DIMH + o] = f2bf(a);
    }
}

// ---------------------------------------------------------------------------
// Generic WMMA GEMM: C(M,N) = A(M,K) @ W(N,K)^T + bias
// grid = (M/16, N/512), block = 256 (8 waves); wave -> 16x64 tile
// ---------------------------------------------------------------------------
template <int OUT_BF>
__global__ __launch_bounds__(256) void k_gemm(const __bf16* __restrict__ A, int lda,
        const __bf16* __restrict__ W, int ldw, const float* __restrict__ bias,
        void* __restrict__ Cv, int ldc, int K) {
    int lane = threadIdx.x & 31, w = threadIdx.x >> 5;
    int m0 = blockIdx.x * 16;
    int n_base = blockIdx.y * 512 + w * 64;
    int r = lane & 15;
    int moff = (lane >> 4) << 3;            // C/D row offset for this half-wave
    const __bf16* arow = A + (size_t)(m0 + r) * lda;
    const __bf16* wrow[4];
#pragma unroll
    for (int t = 0; t < 4; t++) wrow[t] = W + (size_t)(n_base + t * 16 + r) * ldw;
    v8f acc[4] = {};
    for (int k0 = 0; k0 < K; k0 += 32) {
        __builtin_prefetch(arow + k0 + 256, 0, 1);
        v16bf a = frag_a(arow, k0, lane);
#pragma unroll
        for (int t = 0; t < 4; t++) {
            v16bf b = frag_b(wrow[t], k0, lane);
            acc[t] = wmma_bf16(a, b, acc[t]);
        }
    }
#pragma unroll
    for (int t = 0; t < 4; t++) {
        int n = n_base + t * 16 + r;
        float bv = bias ? bias[n] : 0.0f;
#pragma unroll
        for (int v = 0; v < 8; v++) {
            float val = acc[t][v] + bv;
            size_t o = (size_t)(m0 + moff + v) * ldc + n;
            if (OUT_BF) ((__bf16*)Cv)[o] = f2bf(val);
            else        ((float*)Cv)[o]  = val;
        }
    }
}

// ---------------------------------------------------------------------------
// GRU scan on the WMMA pipe. Single block, 8 waves, all 4 batch elements
// packed as rows 0..3 of a 16x512 bf16 LDS tile (rows 4..15 = zero).
// Per step:  gh(16x1536) = h16 @ Whh^T  (8 waves x 12 tiles x 16 k-steps of
// v_wmma_f32_16x16x32_bf16), then f32 gate math with f32 h state.
// xp = x@Wih^T + bih was precomputed by the parallel WMMA GEMM.
// ---------------------------------------------------------------------------
__global__ __launch_bounds__(256) void k_gru_wmma(const float* __restrict__ xp,
        const __bf16* __restrict__ Whh, const float* __restrict__ bhh,
        __bf16* __restrict__ out, int T) {
    __shared__ __bf16 hbf[16][DIMH];   // rows 0..3 = batches, 4..15 = zero
    __shared__ float  h32[BATCH][DIMH];
    __shared__ float  gh[BATCH][H3];
    int tid = threadIdx.x, lane = tid & 31, w = tid >> 5;
    int r = lane & 15, moff = (lane >> 4) << 3;
    for (int i = tid; i < 16 * DIMH; i += 256) hbf[i >> 9][i & 511] = f2bf(0.0f);
    for (int i = tid; i < BATCH * DIMH; i += 256) h32[i >> 9][i & 511] = 0.0f;
    __syncthreads();

    const __bf16* wrow[12];
#pragma unroll
    for (int nt = 0; nt < 12; nt++)
        wrow[nt] = Whh + (size_t)(w * 192 + nt * 16 + r) * DIMH;

    for (int t = 0; t < T; ++t) {
        // ---- recurrent matmul on XDL ----
        v8f acc[12] = {};
        for (int k0 = 0; k0 < DIMH; k0 += 32) {
            v16bf a = frag_a(&hbf[r][0], k0, lane);
#pragma unroll
            for (int nt = 0; nt < 12; nt++) {
                v16bf b = frag_b(wrow[nt], k0, lane);
                acc[nt] = wmma_bf16(a, b, acc[nt]);
            }
        }
        if (moff == 0) {                     // rows 0..3 carry real batches
#pragma unroll
            for (int nt = 0; nt < 12; nt++) {
                int n = w * 192 + nt * 16 + r;
                float bv = bhh[n];
#pragma unroll
                for (int v = 0; v < BATCH; v++) gh[v][n] = acc[nt][v] + bv;
            }
        }
        __syncthreads();

        // ---- gates: 4 batches x 512 dims = 2048 lanes of work ----
#pragma unroll
        for (int q = 0; q < 8; q++) {
            int idx = tid + q * 256;
            int b = idx >> 9, d = idx & 511;
            const float* xrow = xp + ((size_t)b * T + t) * H3;
            float xr = xrow[d], xz = xrow[DIMH + d], xn = xrow[2 * DIMH + d];
            float rr = 1.0f / (1.0f + expf(-(xr + gh[b][d])));
            float zz = 1.0f / (1.0f + expf(-(xz + gh[b][DIMH + d])));
            float nn = tanhf(xn + rr * gh[b][2 * DIMH + d]);
            float hnew = (1.0f - zz) * nn + zz * h32[b][d];
            h32[b][d] = hnew;                 // each (b,d) owned by one thread
            hbf[b][d] = f2bf(hnew);
            out[((size_t)b * T + t) * DIMH + d] = f2bf(hnew);
        }
        __syncthreads();
    }
}

// ---------------------------------------------------------------------------
// Fused final MLP: per block of 16 tokens
//   h0 = conv(emb gather, K=2048) + bc + cond        -> LDS
//   h1 = relu(h0@W1^T+b1), h2 = relu(h1@W2^T+b2)      -> LDS
//   out = log_softmax(h2@Wout^T+bout)                 -> global f32
// ---------------------------------------------------------------------------
__device__ inline void lds_stage(const __bf16 (*src)[DIMH], const __bf16* __restrict__ W,
                                 const float* __restrict__ bias, __bf16 (*dst)[DIMH],
                                 int lane, int w, bool relu) {
    int r = lane & 15, moff = (lane >> 4) << 3;
    int n0 = w * 64;
    v8f acc[4] = {};
    for (int k0 = 0; k0 < DIMH; k0 += 32) {
        v16bf a = frag_a(&src[r][0], k0, lane);
#pragma unroll
        for (int t = 0; t < 4; t++) {
            v16bf b = frag_b(W + (size_t)(n0 + t * 16 + r) * DIMH, k0, lane);
            acc[t] = wmma_bf16(a, b, acc[t]);
        }
    }
#pragma unroll
    for (int t = 0; t < 4; t++) {
        int n = n0 + t * 16 + r;
        float bv = bias[n];
#pragma unroll
        for (int v = 0; v < 8; v++) {
            float val = acc[t][v] + bv;
            if (relu) val = fmaxf(val, 0.0f);
            dst[moff + v][n] = f2bf(val);
        }
    }
}

__global__ __launch_bounds__(256) void k_final(const int* __restrict__ inp,
        const __bf16* __restrict__ emb, const __bf16* __restrict__ WcK,
        const float* __restrict__ bc,
        const __bf16* __restrict__ W1, const float* __restrict__ b1,
        const __bf16* __restrict__ W2, const float* __restrict__ b2,
        const __bf16* __restrict__ Wout, const float* __restrict__ bout,
        const __bf16* __restrict__ cond, float* __restrict__ out) {
    __shared__ __bf16 hA[16][DIMH];
    __shared__ __bf16 hB[16][DIMH];
    __shared__ float  logits[16][256];
    __shared__ int    sidx[32];
    __shared__ float  rstat[16][2];
    int tid = threadIdx.x, lane = tid & 31, w = tid >> 5;
    int gt0 = blockIdx.x * 16;
    int b = gt0 / TTOK, t0 = gt0 % TTOK;
    if (tid < 31) sidx[tid] = inp[(size_t)b * LFULL + 48 + t0 + tid];
    __syncthreads();
    int r = lane & 15, moff = (lane >> 4) << 3;

    // ---- Stage A: conv as K=2048 GEMM with gathered embedding rows ----
    {
        int n0 = w * 64;
        v8f acc[4] = {};
        for (int k0 = 0; k0 < 2048; k0 += 32) {
            int tap = k0 >> 7;                       // constant across this k-step
            int c0  = (k0 + moff) & 127;
            const __bf16* ep = emb + (size_t)sidx[r + tap] * 128 + c0;
            v8bf lo = *(const v8bf*)ep;
            v8bf hi = *(const v8bf*)(ep + 16);
            v16bf a = __builtin_shufflevector(lo, hi, 0,1,2,3,4,5,6,7,8,9,10,11,12,13,14,15);
#pragma unroll
            for (int t = 0; t < 4; t++) {
                v16bf bb = frag_b(WcK + (size_t)(n0 + t * 16 + r) * 2048, k0, lane);
                acc[t] = wmma_bf16(a, bb, acc[t]);
            }
        }
#pragma unroll
        for (int t = 0; t < 4; t++) {
            int n = n0 + t * 16 + r;
            float bv = bc[n];
#pragma unroll
            for (int v = 0; v < 8; v++) {
                int m = moff + v;
                float val = acc[t][v] + bv + bf2f(cond[(size_t)(gt0 + m) * DIMH + n]);
                hA[m][n] = f2bf(val);
            }
        }
    }
    __syncthreads();
    lds_stage(hA, W1, b1, hB, lane, w, true);
    __syncthreads();
    lds_stage(hB, W2, b2, hA, lane, w, true);
    __syncthreads();

    // ---- Stage D: logits = hA @ Wout^T + bout  (N = 256) ----
    {
        int n0 = w * 32;
        v8f acc[2] = {};
        for (int k0 = 0; k0 < DIMH; k0 += 32) {
            v16bf a = frag_a(&hA[r][0], k0, lane);
#pragma unroll
            for (int t = 0; t < 2; t++) {
                v16bf bb = frag_b(Wout + (size_t)(n0 + t * 16 + r) * DIMH, k0, lane);
                acc[t] = wmma_bf16(a, bb, acc[t]);
            }
        }
#pragma unroll
        for (int t = 0; t < 2; t++) {
            int n = n0 + t * 16 + r;
            float bv = bout[n];
#pragma unroll
            for (int v = 0; v < 8; v++) logits[moff + v][n] = acc[t][v] + bv;
        }
    }
    __syncthreads();
    if (tid < 16) {
        float mx = -3.0e38f;
        for (int j = 0; j < 256; j++) mx = fmaxf(mx, logits[tid][j]);
        float se = 0.0f;
        for (int j = 0; j < 256; j++) se += expf(logits[tid][j] - mx);
        rstat[tid][0] = mx;
        rstat[tid][1] = logf(se);
    }
    __syncthreads();
    for (int i = tid; i < 16 * 256; i += 256) {
        int m = i >> 8, n = i & 255;
        out[(size_t)(gt0 + m) * 256 + n] = logits[m][n] - rstat[m][0] - rstat[m][1];
    }
}

// ---------------------------------------------------------------------------
// Host-side orchestration
// ---------------------------------------------------------------------------
extern "C" void kernel_launch(void* const* d_in, const int* in_sizes, int n_in,
                              void* d_out, int out_size, void* d_ws, size_t ws_size,
                              hipStream_t stream) {
    const int*   inp  = (const int*)  d_in[0];
    const float* emb  = (const float*)d_in[2];
    const float* Wc   = (const float*)d_in[3];
    const float* bc   = (const float*)d_in[4];
    const float* W1   = (const float*)d_in[5];
    const float* b1   = (const float*)d_in[6];
    const float* W2   = (const float*)d_in[7];
    const float* b2   = (const float*)d_in[8];
    const float* Wout = (const float*)d_in[9];
    const float* bout = (const float*)d_in[10];
    const float* Wint = (const float*)d_in[11];
    const float* bint = (const float*)d_in[12];
    const float* Wiht = (const float*)d_in[13];
    const float* Whht = (const float*)d_in[14];
    const float* biht = (const float*)d_in[15];
    const float* bhht = (const float*)d_in[16];
    const float* Wupt = (const float*)d_in[17];
    const float* bupt = (const float*)d_in[18];
    const float* Winb = (const float*)d_in[19];
    const float* binb = (const float*)d_in[20];
    const float* Wihb = (const float*)d_in[21];
    const float* Whhb = (const float*)d_in[22];
    const float* bihb = (const float*)d_in[23];
    const float* bhhb = (const float*)d_in[24];
    const float* Wupb = (const float*)d_in[25];
    const float* bupb = (const float*)d_in[26];
    (void)in_sizes; (void)n_in; (void)out_size; (void)ws_size;

    char* ws = (char*)d_ws;
    size_t cur = 0;
    auto alloc = [&](size_t bytes) -> char* {
        char* p = ws + cur;
        cur = (cur + bytes + 255) & ~(size_t)255;
        return p;
    };
    __bf16* emb_b    = (__bf16*)alloc(32768u * 2);
    __bf16* WcK_b    = (__bf16*)alloc(512u * 2048u * 2);
    __bf16* W1_b     = (__bf16*)alloc(262144u * 2);
    __bf16* W2_b     = (__bf16*)alloc(262144u * 2);
    __bf16* Wout_b   = (__bf16*)alloc(131072u * 2);
    __bf16* Wiht_b   = (__bf16*)alloc(786432u * 2);
    __bf16* Whht_b   = (__bf16*)alloc(786432u * 2);
    __bf16* Wupt_b   = (__bf16*)alloc(1048576u * 2);
    __bf16* Wihb_b   = (__bf16*)alloc(786432u * 2);
    __bf16* Whhb_b   = (__bf16*)alloc(786432u * 2);
    __bf16* Wupb_b   = (__bf16*)alloc(4194304u * 2);
    float*  hf       = (float*) alloc((size_t)BATCH * SFRM * 13 * 4);
    __bf16* xtop_b   = (__bf16*)alloc((size_t)BATCH * TTOP * DIMH * 2);
    float*  xp_top   = (float*) alloc((size_t)BATCH * TTOP * H3 * 4);
    __bf16* outtop_b = (__bf16*)alloc((size_t)BATCH * TTOP * DIMH * 2);
    float*  condtop  = (float*) alloc((size_t)BATCH * TBOT * DIMH * 4);
    __bf16* xbot_b   = (__bf16*)alloc((size_t)BATCH * TBOT * DIMH * 2);
    float*  xp_bot   = (float*) alloc((size_t)BATCH * TBOT * H3 * 4);
    __bf16* outbot_b = (__bf16*)alloc((size_t)BATCH * TBOT * DIMH * 2);
    __bf16* cond_b   = (__bf16*)alloc((size_t)BATCH * TTOK * DIMH * 2);

    // --- weight conversion ---
    k_cvt<<<128,  256, 0, stream>>>(emb,  emb_b,  32768);
    k_cvt_wc<<<1024, 256, 0, stream>>>(Wc, WcK_b);
    k_cvt<<<512,  256, 0, stream>>>(W1,   W1_b,   262144);
    k_cvt<<<512,  256, 0, stream>>>(W2,   W2_b,   262144);
    k_cvt<<<512,  256, 0, stream>>>(Wout, Wout_b, 131072);
    k_cvt<<<1024, 256, 0, stream>>>(Wiht, Wiht_b, 786432);
    k_cvt<<<1024, 256, 0, stream>>>(Whht, Whht_b, 786432);
    k_cvt<<<1024, 256, 0, stream>>>(Wupt, Wupt_b, 1048576);
    k_cvt<<<1024, 256, 0, stream>>>(Wihb, Wihb_b, 786432);
    k_cvt<<<1024, 256, 0, stream>>>(Whhb, Whhb_b, 786432);
    k_cvt<<<2048, 256, 0, stream>>>(Wupb, Wupb_b, 4194304);

    // --- features + top path ---
    k_features<<<(BATCH * SFRM * 13 + 255) / 256, 256, 0, stream>>>(inp, hf);
    k_xtop<<<BATCH * TTOP, 256, 0, stream>>>(inp, hf, Wint, bint, xtop_b);
    k_gemm<0><<<dim3(BATCH * TTOP / 16, 3), 256, 0, stream>>>(
        xtop_b, DIMH, Wiht_b, DIMH, biht, xp_top, H3, DIMH);
    k_gru_wmma<<<1, 256, 0, stream>>>(xp_top, Whht_b, bhht, outtop_b, TTOP);
    k_gemm<0><<<dim3(BATCH * TTOP / 16, 4), 256, 0, stream>>>(
        outtop_b, DIMH, Wupt_b, DIMH, bupt, condtop, 2048, DIMH);

    // --- bottom path ---
    k_xbot<<<BATCH * TBOT, 256, 0, stream>>>(inp, hf, Winb, binb, condtop, xbot_b);
    k_gemm<0><<<dim3(BATCH * TBOT / 16, 3), 256, 0, stream>>>(
        xbot_b, DIMH, Wihb_b, DIMH, bihb, xp_bot, H3, DIMH);
    k_gru_wmma<<<1, 256, 0, stream>>>(xp_bot, Whhb_b, bhhb, outbot_b, TBOT);
    k_gemm<1><<<dim3(BATCH * TBOT / 16, 16), 256, 0, stream>>>(
        outbot_b, DIMH, Wupb_b, DIMH, bupb, cond_b, 8192, DIMH);

    // --- fused conv + MLP + log_softmax ---
    k_final<<<BATCH * TTOK / 16, 256, 0, stream>>>(
        inp, emb_b, WcK_b, bc, W1_b, b1, W2_b, b2, Wout_b, bout, cond_b,
        (float*)d_out);
}
